// FGG_51591147160131
// MI455X (gfx1250) — matrix operand order repaired and checked by reading, not compiled, not run on hardware
//
#include <hip/hip_runtime.h>

typedef __attribute__((ext_vector_type(16))) __bf16 v16bf;
typedef __attribute__((ext_vector_type(8)))  float  v8f;
typedef unsigned short u16;

#define F_ACC   1
#define F_BIAS  2
#define F_ELU   4
#define F_WF32  8
#define F_WBF16 16

static __device__ __forceinline__ u16 f2bf(float f) {
  union { float f; unsigned u; } v; v.f = f;
  unsigned r = v.u + 0x7FFFu + ((v.u >> 16) & 1u);   // round-to-nearest-even
  return (u16)(r >> 16);
}

union Frag { v16bf v; uint4 q[2]; };

// ---------------- f32 -> bf16 conversion ----------------
__global__ __launch_bounds__(256)
void f32_to_bf16_kernel(const float* __restrict__ in, u16* __restrict__ out, size_t n) {
  size_t i = ((size_t)blockIdx.x * 256 + threadIdx.x) * 4;
  if (i + 4 <= n) {
    float4 v = *(const float4*)(in + i);
    ushort4 o;
    o.x = f2bf(v.x); o.y = f2bf(v.y); o.z = f2bf(v.z); o.w = f2bf(v.w);
    *(ushort4*)(out + i) = o;
  } else {
    for (; i < n; ++i) out[i] = f2bf(in[i]);
  }
}

// ---------------- deterministic BN reduction ----------------
__global__ __launch_bounds__(256)
void bn_partial_kernel(const float* __restrict__ h, float* __restrict__ psum,
                       float* __restrict__ psq, int rowsPerChunk, int F) {
  int c = blockIdx.x * 256 + threadIdx.x;
  if (c >= F) return;
  int r0 = blockIdx.y * rowsPerChunk;
  float s = 0.f, s2 = 0.f;
  for (int r = 0; r < rowsPerChunk; ++r) {
    float v = h[(size_t)(r0 + r) * F + c];
    s += v; s2 += v * v;
  }
  psum[(size_t)blockIdx.y * F + c] = s;
  psq [(size_t)blockIdx.y * F + c] = s2;
}

__global__ __launch_bounds__(256)
void bn_final_kernel(const float* __restrict__ psum, const float* __restrict__ psq,
                     float* __restrict__ mean, float* __restrict__ invstd,
                     int F, int rows, int chunks) {
  int c = blockIdx.x * 256 + threadIdx.x;
  if (c >= F) return;
  float s = 0.f, s2 = 0.f;
  for (int p = 0; p < chunks; ++p) { s += psum[(size_t)p * F + c]; s2 += psq[(size_t)p * F + c]; }
  float m = s / (float)rows;
  float var = s2 / (float)rows - m * m;
  mean[c] = m;
  invstd[c] = rsqrtf(var + 1e-5f);
}

__global__ __launch_bounds__(256)
void bn_apply_kernel(const float* __restrict__ h, const float* __restrict__ mean,
                     const float* __restrict__ invstd, const float* __restrict__ g,
                     const float* __restrict__ be, u16* __restrict__ outb,
                     size_t total, int F) {
  size_t i = (size_t)blockIdx.x * 256 + threadIdx.x;
  if (i >= total) return;
  int c = (int)(i % (size_t)F);
  float v = (h[i] - mean[c]) * invstd[c] * g[c] + be[c];
  v = v > 0.f ? v : __expf(v) - 1.f;                 // ELU
  outb[i] = f2bf(v);
}

// ---------------- split-K partial combine (deterministic) ----------------
__global__ __launch_bounds__(256)
void combine_kernel(const float* __restrict__ P, float* __restrict__ out, size_t MN, int S) {
  size_t i = (size_t)blockIdx.x * 256 + threadIdx.x;
  if (i >= MN) return;
  float s = 0.f;
  for (int p = 0; p < S; ++p) s += P[(size_t)p * MN + i];
  out[i] = s;
}

// ---------------- bf16 WMMA GEMM, 128x128 tile, BK=32, ping-pong LDS ------
// C[M,N](f32) (+)= A[M,K](bf16) @ B[K,N](bf16); FLAGS compile-time epilogue.
// Split-K: blockIdx.z processes K/gridDim.z and writes to C + z*M*N.
// 256 threads = 8 wave32; wave grid 4x2; each wave owns 2x4 tiles of 16x16.
#define LDA 40   // padded LDS stride in shorts (80 B: 16B-aligned fragments)

template <int FLAGS>
__global__ __launch_bounds__(256)
void gemm_bf16_kernel(const u16* __restrict__ A, const u16* __restrict__ B,
                      float* __restrict__ C, u16* __restrict__ Cbf,
                      const float* __restrict__ bias, int M, int N, int K) {
  __shared__ __align__(16) u16 lA[2][128 * LDA];
  __shared__ __align__(16) u16 lBt[2][128 * LDA];   // [n][k] transposed

  const int tid  = threadIdx.x;
  const int lane = tid & 31;
  const int wave = tid >> 5;
  const int wr = wave >> 1, wc = wave & 1;
  const int rowBase = blockIdx.y * 128;
  const int colBase = blockIdx.x * 128;

  // staging assignments
  const int arow0 = tid >> 2, aseg = tid & 3;            // A: rows tid/4 and +64
  const int brp = tid >> 4, bseg = tid & 15;             // B: k-row pair, 8-col seg

  v8f acc[2][4];
#pragma unroll
  for (int mi = 0; mi < 2; ++mi)
#pragma unroll
    for (int ni = 0; ni < 4; ++ni)
#pragma unroll
      for (int i = 0; i < 8; ++i) acc[mi][ni][i] = 0.f;

  const int kh = lane >> 4;        // lane K-half per 16-bit operand layout
  const int ml = lane & 15;

  const int Kc = K / gridDim.z;
  const int kBegin = blockIdx.z * Kc;
  const int nk = Kc / 32;

  uint4 pa0, pa1, pb0, pb1;
  auto fetch = [&](int k0) {
    pa0 = *(const uint4*)(A + (size_t)(rowBase + arow0) * K + k0 + aseg * 8);
    pa1 = *(const uint4*)(A + (size_t)(rowBase + arow0 + 64) * K + k0 + aseg * 8);
    pb0 = *(const uint4*)(B + (size_t)(k0 + 2 * brp) * N + colBase + bseg * 8);
    pb1 = *(const uint4*)(B + (size_t)(k0 + 2 * brp + 1) * N + colBase + bseg * 8);
  };
  auto stage = [&](int buf) {
    *(uint4*)(&lA[buf][arow0 * LDA + aseg * 8]) = pa0;
    *(uint4*)(&lA[buf][(arow0 + 64) * LDA + aseg * 8]) = pa1;
    const u16* s0 = (const u16*)&pb0;
    const u16* s1 = (const u16*)&pb1;
#pragma unroll
    for (int j = 0; j < 8; ++j) {
      unsigned packed = (unsigned)s0[j] | ((unsigned)s1[j] << 16);  // (k, k+1)
      *(unsigned*)(&lBt[buf][(bseg * 8 + j) * LDA + 2 * brp]) = packed;
    }
  };

  fetch(kBegin);
  stage(0);
  __syncthreads();

  for (int it = 0; it < nk; ++it) {
    const int buf = it & 1;
    const bool more = (it + 1) < nk;
    if (more) fetch(kBegin + (it + 1) * 32);   // overlap global latency w/ compute

    Frag a[2], b[4];
#pragma unroll
    for (int mi = 0; mi < 2; ++mi) {
      int ar = wr * 32 + mi * 16 + ml;
      a[mi].q[0] = *(const uint4*)(&lA[buf][ar * LDA + kh * 8]);       // K 0..7 / 8..15
      a[mi].q[1] = *(const uint4*)(&lA[buf][ar * LDA + 16 + kh * 8]);  // K 16..23 / 24..31
    }
#pragma unroll
    for (int ni = 0; ni < 4; ++ni) {
      int bn = wc * 64 + ni * 16 + ml;
      b[ni].q[0] = *(const uint4*)(&lBt[buf][bn * LDA + kh * 8]);
      b[ni].q[1] = *(const uint4*)(&lBt[buf][bn * LDA + 16 + kh * 8]);
    }
#pragma unroll
    for (int mi = 0; mi < 2; ++mi)
#pragma unroll
      for (int ni = 0; ni < 4; ++ni)
        acc[mi][ni] = __builtin_amdgcn_wmma_f32_16x16x32_bf16(
            false, a[mi].v, false, b[ni].v, (short)0, acc[mi][ni], false, false);

    if (more) stage(buf ^ 1);
    __syncthreads();
  }

  // epilogue; C/D layout: VGPR i -> row (lane/16)*8 + i, col lane%16
  const int rl = lane >> 4, cl = lane & 15;
  const size_t zoff = (size_t)blockIdx.z * M * N;
#pragma unroll
  for (int ni = 0; ni < 4; ++ni) {
    const int col = colBase + wc * 64 + ni * 16 + cl;
    float bv = 0.f;
    if constexpr (FLAGS & F_BIAS) bv = bias[col];
#pragma unroll
    for (int mi = 0; mi < 2; ++mi) {
      const int row0 = rowBase + wr * 32 + mi * 16 + rl * 8;
#pragma unroll
      for (int i = 0; i < 8; ++i) {
        size_t o = zoff + (size_t)(row0 + i) * N + col;
        float v = acc[mi][ni][i];
        if constexpr (FLAGS & F_ACC)  v += C[o];
        if constexpr (FLAGS & F_BIAS) v += bv;
        if constexpr (FLAGS & F_ELU)  v = v > 0.f ? v : __expf(v) - 1.f;
        if constexpr (FLAGS & F_WF32) C[o] = v;
        if constexpr (FLAGS & F_WBF16) Cbf[o] = f2bf(v);
      }
    }
  }
}

extern "C" void kernel_launch(void* const* d_in, const int* in_sizes, int n_in,
                              void* d_out, int out_size, void* d_ws, size_t ws_size,
                              hipStream_t stream) {
  (void)in_sizes; (void)n_in; (void)out_size; (void)ws_size;

  const float* features = (const float*)d_in[0];   // [8192,2048]
  const float* Aadj     = (const float*)d_in[1];   // [2,8192,8192]
  const float* Wd       = (const float*)d_in[2];   // [2048,1024]
  const float* bd       = (const float*)d_in[3];
  const float* W1       = (const float*)d_in[4];   // [2,1024,512]
  const float* b1       = (const float*)d_in[5];
  const float* Wr1      = (const float*)d_in[6];   // [1024,512]
  const float* W2       = (const float*)d_in[7];   // [2,512,256]
  const float* b2       = (const float*)d_in[8];
  const float* Wr2      = (const float*)d_in[9];   // [512,256]
  const float* W3       = (const float*)d_in[10];  // [2,256,128]
  const float* b3       = (const float*)d_in[11];
  const float* Wr3      = (const float*)d_in[12];  // [256,128]
  const float* g1       = (const float*)d_in[13];
  const float* be1      = (const float*)d_in[14];
  const float* g2       = (const float*)d_in[15];
  const float* be2      = (const float*)d_in[16];
  float* out = (float*)d_out;                      // [8192,128]

  const int N = 8192, Fin = 2048, Fd = 1024, F1 = 512, F2 = 256, F3 = 128;
  const int BNCH = 32;   // BN row chunks

  // ---- carve scratch ----
  char* ws = (char*)d_ws;
  size_t off = 0;
  auto alloc = [&](size_t bytes) -> void* {
    void* p = ws + off;
    off = (off + bytes + 255) & ~(size_t)255;
    return p;
  };
  u16*   Abf   = (u16*)alloc((size_t)2 * N * N * 2);       // 256 MB
  u16*   featb = (u16*)alloc((size_t)N * Fin * 2);         // 32 MB
  u16*   Wdb   = (u16*)alloc((size_t)Fin * Fd * 2);
  u16*   W1b   = (u16*)alloc((size_t)2 * Fd * F1 * 2);
  u16*   Wr1b  = (u16*)alloc((size_t)Fd * F1 * 2);
  u16*   W2b   = (u16*)alloc((size_t)2 * F1 * F2 * 2);
  u16*   Wr2b  = (u16*)alloc((size_t)F1 * F2 * 2);
  u16*   W3b   = (u16*)alloc((size_t)2 * F2 * F3 * 2);
  u16*   Wr3b  = (u16*)alloc((size_t)F2 * F3 * 2);
  u16*   xbf   = (u16*)alloc((size_t)N * Fd * 2);          // 16 MB
  u16*   supb  = (u16*)alloc((size_t)2 * N * F1 * 2);      // 16 MB
  float* hf    = (float*)alloc((size_t)N * F1 * 4);        // 16 MB
  u16*   hnb   = (u16*)alloc((size_t)N * F1 * 2);          // 8 MB
  float* part  = (float*)alloc((size_t)8 * N * F3 * 4);    // 32 MB split-K partials
  float* bnp   = (float*)alloc((size_t)2 * BNCH * 1024 * 4);
  float* stats = (float*)alloc((size_t)2 * 1024 * 4);
  float* psum = bnp, *psq = bnp + (size_t)BNCH * 1024;
  float* mean = stats, *invstd = stats + 1024;
  u16* sup0 = supb;
  u16* sup1 = supb + (size_t)N * F1;

  auto conv = [&](const float* s, u16* d, size_t n) {
    f32_to_bf16_kernel<<<dim3((unsigned)((n + 1023) / 1024)), 256, 0, stream>>>(s, d, n);
  };
  auto bn = [&](float* h, const float* g, const float* be, u16* ob, int F) {
    bn_partial_kernel<<<dim3((F + 255) / 256, BNCH), 256, 0, stream>>>(h, psum, psq, N / BNCH, F);
    bn_final_kernel<<<dim3((F + 255) / 256), 256, 0, stream>>>(psum, psq, mean, invstd, F, N, BNCH);
    size_t tot = (size_t)N * F;
    bn_apply_kernel<<<dim3((unsigned)((tot + 255) / 256)), 256, 0, stream>>>(h, mean, invstd, g, be, ob, tot, F);
  };
#define GEMM(FL, A_, B_, C_, Cb_, bias_, M_, N_, K_, Z_)                                   \
  gemm_bf16_kernel<(FL)><<<dim3((N_) / 128, (M_) / 128, (Z_)), 256, 0, stream>>>(          \
      (A_), (B_), (C_), (Cb_), (bias_), (M_), (N_), (K_))

  // ---- convert all GEMM operands to bf16 ----
  conv(features, featb, (size_t)N * Fin);
  conv(Aadj, Abf, (size_t)2 * N * N);
  conv(Wd, Wdb, (size_t)Fin * Fd);
  conv(W1, W1b, (size_t)2 * Fd * F1);
  conv(Wr1, Wr1b, (size_t)Fd * F1);
  conv(W2, W2b, (size_t)2 * F1 * F2);
  conv(Wr2, Wr2b, (size_t)F1 * F2);
  conv(W3, W3b, (size_t)2 * F2 * F3);
  conv(Wr3, Wr3b, (size_t)F2 * F3);

  // ---- x = elu(features @ Wd + bd)  (bf16 out) ----
  GEMM(F_BIAS | F_ELU | F_WBF16, featb, Wdb, (float*)nullptr, xbf, bd, N, Fd, Fin, 1);

  // ---- layer 1: 1024 -> 512 (256 blocks/agg: no split needed) ----
  GEMM(F_WBF16, xbf, W1b, (float*)nullptr, sup0, (const float*)nullptr, N, F1, Fd, 1);
  GEMM(F_WBF16, xbf, W1b + (size_t)Fd * F1, (float*)nullptr, sup1, (const float*)nullptr, N, F1, Fd, 1);
  GEMM(F_WF32, Abf, sup0, hf, (u16*)nullptr, (const float*)nullptr, N, F1, N, 1);
  GEMM(F_ACC | F_WF32, Abf + (size_t)N * N, sup1, hf, (u16*)nullptr, (const float*)nullptr, N, F1, N, 1);
  GEMM(F_ACC | F_BIAS | F_WF32, xbf, Wr1b, hf, (u16*)nullptr, b1, N, F1, Fd, 1);
  bn(hf, g1, be1, hnb, F1);

  // ---- layer 2: 512 -> 256 (split-K x2 per edge type for occupancy) ----
  GEMM(F_WBF16, hnb, W2b, (float*)nullptr, sup0, (const float*)nullptr, N, F2, F1, 1);
  GEMM(F_WBF16, hnb, W2b + (size_t)F1 * F2, (float*)nullptr, sup1, (const float*)nullptr, N, F2, F1, 1);
  {
    size_t MN = (size_t)N * F2;
    GEMM(F_WF32, Abf, sup0, part, (u16*)nullptr, (const float*)nullptr, N, F2, N, 2);
    GEMM(F_WF32, Abf + (size_t)N * N, sup1, part + 2 * MN, (u16*)nullptr, (const float*)nullptr, N, F2, N, 2);
    combine_kernel<<<dim3((unsigned)((MN + 255) / 256)), 256, 0, stream>>>(part, hf, MN, 4);
  }
  GEMM(F_ACC | F_BIAS | F_WF32, hnb, Wr2b, hf, (u16*)nullptr, b2, N, F2, F1, 1);
  bn(hf, g2, be2, hnb, F2);

  // ---- layer 3: 256 -> 128 (split-K x4 per edge type), into d_out ----
  GEMM(F_WBF16, hnb, W3b, (float*)nullptr, sup0, (const float*)nullptr, N, F3, F2, 1);
  GEMM(F_WBF16, hnb, W3b + (size_t)F2 * F3, (float*)nullptr, sup1, (const float*)nullptr, N, F3, F2, 1);
  {
    size_t MN = (size_t)N * F3;
    GEMM(F_WF32, Abf, sup0, part, (u16*)nullptr, (const float*)nullptr, N, F3, N, 4);
    GEMM(F_WF32, Abf + (size_t)N * N, sup1, part + 4 * MN, (u16*)nullptr, (const float*)nullptr, N, F3, N, 4);
    combine_kernel<<<dim3((unsigned)((MN + 255) / 256)), 256, 0, stream>>>(part, out, MN, 8);
  }
  GEMM(F_ACC | F_BIAS | F_WF32, hnb, Wr3b, out, (u16*)nullptr, b3, N, F3, F2, 1);
#undef GEMM
}
